// RBFInterpolator_77386720739790
// MI455X (gfx1250) — compile-verified, matrix operand order checked by reading, not run on hardware
//
#include <hip/hip_runtime.h>

typedef __attribute__((ext_vector_type(16))) _Float16 v16h;
typedef __attribute__((ext_vector_type(8)))  float    v8f;
typedef int v4i_vs __attribute__((vector_size(16)));   // matches builtin param

#define QTOT 16384
#define PTOT 4096
#define STOT 8
#define RTOT 4

#define WAVES_PER_BLOCK 8
#define PCHUNK 1024          // p-points staged per block
#define CF_STRIDE 1032       // padded f16 stride (bank spread)

#if __has_builtin(__builtin_amdgcn_global_load_async_to_lds_b128) && \
    __has_builtin(__builtin_amdgcn_s_wait_asynccnt)
#define USE_ASYNC_LDS 1
#else
#define USE_ASYNC_LDS 0
#endif

// ---------------------------------------------------------------------------
// Kernel 1: initialize out with the polynomial block:
//   out[q][s] = sum_r prod_k xhat[q][k]^powers[r][k] * coeffs[P+r][s]
// (writes, does not add -> no dependence on poisoned d_out)
// ---------------------------------------------------------------------------
__global__ void rbf_poly_init(const float* __restrict__ x,
                              const float* __restrict__ coeffs,
                              const float* __restrict__ shift,
                              const float* __restrict__ scale,
                              const int*   __restrict__ powers,
                              float*       __restrict__ out) {
    int q = blockIdx.x * blockDim.x + threadIdx.x;
    if (q >= QTOT) return;
    float xh[3];
    #pragma unroll
    for (int k = 0; k < 3; ++k) xh[k] = (x[q * 3 + k] - shift[k]) / scale[k];
    float acc[STOT];
    #pragma unroll
    for (int s = 0; s < STOT; ++s) acc[s] = 0.f;
    for (int r = 0; r < RTOT; ++r) {
        float t = 1.f;
        #pragma unroll
        for (int k = 0; k < 3; ++k) {
            int e = powers[r * 3 + k];
            for (int i = 0; i < e; ++i) t *= xh[k];
        }
        #pragma unroll
        for (int s = 0; s < STOT; ++s)
            acc[s] += t * coeffs[(PTOT + r) * STOT + s];
    }
    #pragma unroll
    for (int s = 0; s < STOT; ++s) out[q * STOT + s] = acc[s];
}

// ---------------------------------------------------------------------------
// Kernel 2: out[q][s] += sum_p (r^2 log r) * coeffs[p][s]  via WMMA.
//   r^2 log r = (0.5*ln2) * r2 * log2(r2)   -- no sqrt needed.
// Each wave: one 16-row q tile x full S (padded to 16), over a 1024-p slice.
// y slice arrives via GLOBAL_LOAD_ASYNC_TO_LDS_B128 (ASYNCcnt path), then a
// one-time LDS transpose to SoA so the hot loop reads pure ds_load_b128.
// ---------------------------------------------------------------------------
__global__ __launch_bounds__(256) void
rbf_tps_wmma(const float* __restrict__ x,
             const float* __restrict__ y,
             const float* __restrict__ coeffs,
             float*       __restrict__ out) {
    __shared__ __align__(16) float ys_aos[PCHUNK * 3];  // async landing (12 KB)
    __shared__ __align__(16) float ysx[PCHUNK];         // SoA components
    __shared__ __align__(16) float ysy[PCHUNK];
    __shared__ __align__(16) float ysz[PCHUNK];
    __shared__ _Float16 cf[16 * CF_STRIDE];             // coeffs^T f16, 16 cols

    const int tid = threadIdx.x;
    const int p0  = blockIdx.y * PCHUNK;

    // --- Stage the y slice (contiguous 12 KB) into LDS ---------------------
#if USE_ASYNC_LDS
    {
        const float4* gsrc = (const float4*)(y + (size_t)p0 * 3);
        float4*       ldst = (float4*)ys_aos;
        #pragma unroll
        for (int t = 0; t < 3; ++t) {
            int idx = tid + 256 * t;   // 768 float4 total
            __builtin_amdgcn_global_load_async_to_lds_b128(
                (__attribute__((address_space(1))) v4i_vs*)(gsrc + idx),
                (__attribute__((address_space(3))) v4i_vs*)(ldst + idx),
                0, 0);
        }
    }
#else
    {
        const float4* gsrc = (const float4*)(y + (size_t)p0 * 3);
        float4*       ldst = (float4*)ys_aos;
        #pragma unroll
        for (int t = 0; t < 3; ++t) {
            int idx = tid + 256 * t;
            ldst[idx] = gsrc[idx];
        }
    }
#endif

    // --- Stage coeffs^T as f16 (needs convert+transpose+pad: VALU path) ----
    for (int i = tid; i < 16 * PCHUNK; i += 256) {
        int n = i >> 10;
        int p = i & (PCHUNK - 1);
        float v = (n < STOT) ? coeffs[(p0 + p) * STOT + n] : 0.f;
        cf[n * CF_STRIDE + p] = (_Float16)v;
    }

#if USE_ASYNC_LDS
    __builtin_amdgcn_s_wait_asynccnt(0);
#endif
    __syncthreads();

    // --- One-time AoS -> SoA transpose inside LDS --------------------------
    for (int i = tid; i < PCHUNK; i += 256) {
        ysx[i] = ys_aos[i * 3 + 0];
        ysy[i] = ys_aos[i * 3 + 1];
        ysz[i] = ys_aos[i * 3 + 2];
    }
    __syncthreads();

    const int wave = tid >> 5;
    const int lane = tid & 31;
    const int m    = lane & 15;   // A row / B-D column
    const int h    = lane >> 4;   // half-wave: selects K subsets
    const int qt   = blockIdx.x * WAVES_PER_BLOCK + wave;
    const int q    = qt * 16 + m;

    const float xq0 = x[q * 3 + 0];
    const float xq1 = x[q * 3 + 1];
    const float xq2 = x[q * 3 + 2];

    v8f acc = {0.f, 0.f, 0.f, 0.f, 0.f, 0.f, 0.f, 0.f};
    const float C = 0.34657359027997264f;  // 0.5 * ln(2)

    #pragma unroll 2
    for (int c = 0; c < PCHUNK / 32; ++c) {
        const int base = c * 32;

        // Build A fragment: K tile in f16, directly in WMMA A layout.
        // element e -> K index: e + (e & 8) + 8*h  (relative to base)
        v16h a;
        #pragma unroll
        for (int g = 0; g < 2; ++g) {
            const int kb = base + 16 * g + 8 * h;   // 8 contiguous p points
            float fx[8], fy[8], fz[8];
            *(float4*)&fx[0] = *(const float4*)&ysx[kb];
            *(float4*)&fx[4] = *(const float4*)&ysx[kb + 4];
            *(float4*)&fy[0] = *(const float4*)&ysy[kb];
            *(float4*)&fy[4] = *(const float4*)&ysy[kb + 4];
            *(float4*)&fz[0] = *(const float4*)&ysz[kb];
            *(float4*)&fz[4] = *(const float4*)&ysz[kb + 4];
            #pragma unroll
            for (int j = 0; j < 8; ++j) {
                float dx = xq0 - fx[j];
                float dy = xq1 - fy[j];
                float dz = xq2 - fz[j];
                float r2 = dx * dx + dy * dy + dz * dz;
                float Kv = (r2 > 0.f) ? (C * r2) * __builtin_amdgcn_logf(r2)
                                      : 0.f;
                a[g * 8 + j] = (_Float16)Kv;
            }
        }

        // Build B fragment: coeffs^T from LDS, same element->K mapping.
        v16h b;
        #pragma unroll
        for (int e = 0; e < 16; e += 2) {
            const int kk = base + e + (e & 8) + 8 * h;
            b[e]     = cf[m * CF_STRIDE + kk];
            b[e + 1] = cf[m * CF_STRIDE + kk + 1];
        }

        acc = __builtin_amdgcn_wmma_f32_16x16x32_f16(
            false, a, false, b, (short)0, acc, false, false);
    }

    // D layout: row = v + 8*h, col = m; only cols < STOT are real.
    if (m < STOT) {
        #pragma unroll
        for (int vv = 0; vv < 8; ++vv) {
            int qrow = qt * 16 + vv + 8 * h;
            atomicAdd(&out[qrow * STOT + m], acc[vv]);
        }
    }
}

// ---------------------------------------------------------------------------
extern "C" void kernel_launch(void* const* d_in, const int* in_sizes, int n_in,
                              void* d_out, int out_size, void* d_ws, size_t ws_size,
                              hipStream_t stream) {
    const float* x      = (const float*)d_in[0];
    const float* y      = (const float*)d_in[1];
    const float* coeffs = (const float*)d_in[2];
    const float* shift  = (const float*)d_in[3];
    const float* scale  = (const float*)d_in[4];
    const int*   powers = (const int*)d_in[5];
    float* out = (float*)d_out;

    // 1) polynomial block writes out
    rbf_poly_init<<<QTOT / 256, 256, 0, stream>>>(x, coeffs, shift, scale,
                                                  powers, out);

    // 2) TPS kernel block accumulates via WMMA
    dim3 grid(QTOT / (16 * WAVES_PER_BLOCK), PTOT / PCHUNK);
    rbf_tps_wmma<<<grid, 256, 0, stream>>>(x, y, coeffs, out);
}
